// FastSpeech2Loss_75677323755896
// MI455X (gfx1250) — compile-verified, matrix operand order checked by reading, not run on hardware
//
#include <hip/hip_runtime.h>
#include <hip/hip_bf16.h>

typedef __attribute__((ext_vector_type(16))) _Float16 v16h;
typedef __attribute__((ext_vector_type(8)))  float    v8f;
typedef int v4i_ __attribute__((vector_size(16)));

#define B_   16
#define S_   128
#define T_   800
#define C_   80
#define KPAD 96     // C padded to multiple of 32 for 16x16x32 WMMA
#define TILE 32     // output tile per workgroup (2x2 waves of 16x16)
#define GAMMA 0.1f
#define INFV 1e9f

// ---- CDNA5 async global->LDS path (guarded; falls back to sync copies) ----
#if defined(__has_builtin)
#  if __has_builtin(__builtin_amdgcn_global_load_async_to_lds_b128) && \
      __has_builtin(__builtin_amdgcn_s_wait_asynccnt)
#    define USE_ASYNC_LDS 1
#  endif
#endif
#ifndef USE_ASYNC_LDS
#  define USE_ASYNC_LDS 0
#endif

__device__ __forceinline__ void copy16_to_lds(const void* g, void* l) {
#if USE_ASYNC_LDS
    __builtin_amdgcn_global_load_async_to_lds_b128(
        (__attribute__((address_space(1))) v4i_*)(unsigned long long)g,
        (__attribute__((address_space(3))) v4i_*)(unsigned int)(unsigned long long)l,
        0, 0);
#else
    *(uint4*)l = *(const uint4*)g;
#endif
}

__device__ __forceinline__ void lds_copy_fence() {
#if USE_ASYNC_LDS
    __builtin_amdgcn_s_wait_asynccnt(0);
#endif
}

__device__ __forceinline__ float sigmoidf(float v) {
    return 1.0f / (1.0f + __expf(-v));
}

// lane/half -> K mapping for the 16-bit A-matrix 16x32 fragment (ISA 7.12.2)
__device__ __forceinline__ int kmap(int lane, int h) {
    int v = h >> 1;
    int base = (v < 4) ? (v * 2) : (16 + (v - 4) * 2);
    return base + (h & 1) + ((lane >> 4) << 3);
}

// ---------------------------------------------------------------------------
// Kernel 0: one-shot sigmoid + f16 pack + row norms.
// 3 tensors x 16 batches x 800 rows; one wave per row, lane covers 3 columns.
// Xh[2][16][800][96], Yh[16][800][96], xn[2][16][800], yn[16][800]
// ---------------------------------------------------------------------------
__global__ __launch_bounds__(128)
void prep_kernel(const float* __restrict__ mel,
                 const float* __restrict__ mel_postnet,
                 const float* __restrict__ mel_target,
                 const unsigned char* __restrict__ pred_mel_mask,
                 _Float16* __restrict__ Xh, _Float16* __restrict__ Yh,
                 float* __restrict__ xn, float* __restrict__ yn) {
    const int wave = threadIdx.x >> 5;
    const int lane = threadIdx.x & 31;
    const int rid  = blockIdx.x * 4 + wave;          // 0 .. 3*16*800-1
    const int t    = rid / (B_ * T_);                // tensor: 0=mel,1=postnet,2=target
    const int rem  = rid % (B_ * T_);
    const int b    = rem / T_;
    const int row  = rem % T_;

    const float* src = (t == 0) ? mel : (t == 1) ? mel_postnet : mel_target;
    const bool masked = (t == 0) && (pred_mel_mask[b * T_ + row] != 0);

    _Float16* dst = (t < 2)
        ? Xh + ((size_t)(t * B_ + b) * T_ + row) * KPAD
        : Yh + ((size_t)b * T_ + row) * KPAD;

    float nrm = 0.0f;
    #pragma unroll
    for (int c = 0; c < 3; ++c) {
        int col = lane + 32 * c;
        float v = 0.0f;
        if (col < C_) {
            float raw = masked ? 0.0f : src[((size_t)b * T_ + row) * C_ + col];
            v = sigmoidf(raw);
        }
        dst[col] = (_Float16)v;
        nrm += v * v;
    }
    #pragma unroll
    for (int off = 16; off > 0; off >>= 1) nrm += __shfl_xor(nrm, off, 32);
    if (lane == 0) {
        if (t < 2) xn[(t * B_ + b) * T_ + row] = nrm;
        else       yn[(size_t)b * T_ + row]    = nrm;
    }
}

// ---------------------------------------------------------------------------
// Kernel 1: D[mi][b][i][j] = xn[i] + yn[j] - 2 x_i.y_j  via WMMA.
// Tiles are contiguous 6144-B blocks -> async B128 copies into LDS.
// grid (25, 25, 32), block 128 (4 waves), v_wmma_f32_16x16x32_f16 x3 per wave.
// ---------------------------------------------------------------------------
__global__ __launch_bounds__(128)
void cost_kernel(const _Float16* __restrict__ Xh, const _Float16* __restrict__ Yh,
                 const float* __restrict__ xn, const float* __restrict__ yn,
                 float* __restrict__ D) {
    __shared__ _Float16 Xs[TILE][KPAD];
    __shared__ _Float16 Ys[TILE][KPAD];
    __shared__ float xx[TILE];
    __shared__ float yy[TILE];

    const int i0 = blockIdx.x * TILE;
    const int j0 = blockIdx.y * TILE;
    const int bz = blockIdx.z;
    const int b  = bz >> 1;
    const int mi = bz & 1;
    const int tid = threadIdx.x;

    // contiguous tile blocks: 32 rows * 192 B = 6144 B = 384 x b128 each
    const char* gx = (const char*)(Xh + ((size_t)(mi * B_ + b) * T_ + i0) * KPAD);
    const char* gy = (const char*)(Yh + ((size_t)b * T_ + j0) * KPAD);
    char* lx = (char*)&Xs[0][0];
    char* ly = (char*)&Ys[0][0];
    #pragma unroll
    for (int c = tid; c < 384; c += 128) {
        copy16_to_lds(gx + c * 16, lx + c * 16);
        copy16_to_lds(gy + c * 16, ly + c * 16);
    }
    if (tid < 64) {
        int r = tid & 31;
        if (tid < 32) xx[r] = xn[(mi * B_ + b) * T_ + i0 + r];
        else          yy[r] = yn[(size_t)b * T_ + j0 + r];
    }
    lds_copy_fence();
    __syncthreads();

    const int wave = tid >> 5;
    const int lane = tid & 31;
    const int wm = (wave >> 1) * 16;
    const int wn = (wave & 1) * 16;

    v8f acc = {};
    #pragma unroll
    for (int kk = 0; kk < KPAD; kk += 32) {
        v16h a, bfrag;
        int am = wm + (lane & 15);
        int bn = wn + (lane & 15);
        #pragma unroll
        for (int h = 0; h < 16; ++h) {
            int k = kmap(lane, h) + kk;
            a[h]     = Xs[am][k];
            bfrag[h] = Ys[bn][k];
        }
        acc = __builtin_amdgcn_wmma_f32_16x16x32_f16(
            false, a, false, bfrag, (short)0, acc, false, false);
    }

    // C/D layout: lanes 0-15 -> N=lane, M=v ; lanes 16-31 -> N=lane-16, M=v+8
    const int n    = lane & 15;
    const int mofs = (lane >> 4) << 3;
    float* Dbase = D + ((size_t)(mi * B_ + b) * T_) * T_;
    #pragma unroll
    for (int v = 0; v < 8; ++v) {
        int m  = v + mofs;
        int gi = i0 + wm + m;
        int gj = j0 + wn + n;
        Dbase[(size_t)gi * T_ + gj] = xx[wm + m] + yy[wn + n] - 2.0f * acc[v];
    }
}

// ---------------------------------------------------------------------------
// Kernel 2: soft-DTW anti-diagonal sweep. 32 blocks (batch x matrix),
// 800 threads, 3 rotating LDS diagonals, 1 barrier per diagonal.
// ---------------------------------------------------------------------------
__global__ __launch_bounds__(T_)
void dtw_kernel(const float* __restrict__ D, float* __restrict__ dtw_res) {
    __shared__ float R[3][T_];
    const int bz = blockIdx.x;
    const int i = threadIdx.x;
    const float* Dbase = D + ((size_t)bz * T_) * T_;

    R[0][i] = INFV;
    R[1][i] = INFV;
    __syncthreads();

    const float inv_g = 1.0f / GAMMA;
    for (int p = 0; p < 2 * T_ - 1; ++p) {
        const float* r2 = R[p % 3];
        const float* r1 = R[(p + 1) % 3];
        float*       rc = R[(p + 2) % 3];

        int jj = p - i;
        bool valid = (jj >= 0) && (jj < T_);
        float Dd = valid ? Dbase[(size_t)i * T_ + jj] : 0.0f;
        if (valid && jj + 1 < T_)
            __builtin_prefetch(&Dbase[(size_t)i * T_ + jj + 1], 0, 0);

        float up   = (i > 0) ? r1[i - 1] : INFV;
        float diag = (i > 0) ? r2[i - 1] : ((p == 0) ? 0.0f : INFV);
        float left = r1[i];

        float mmin = fminf(up, fminf(left, diag));
        float s = __expf(-(up   - mmin) * inv_g)
                + __expf(-(left - mmin) * inv_g)
                + __expf(-(diag - mmin) * inv_g);
        float r = Dd + mmin - GAMMA * __logf(s);

        rc[i] = valid ? r : INFV;
        __syncthreads();
    }

    if (i == T_ - 1) {
        int lastp = 2 * T_ - 2;
        dtw_res[bz] = R[(lastp + 2) % 3][T_ - 1];
    }
}

// ---------------------------------------------------------------------------
// Kernel 3: d_loss + means -> (mel_loss, mel_postnet_loss, d_loss)
// ---------------------------------------------------------------------------
__global__ __launch_bounds__(32)
void finalize_kernel(const float* __restrict__ d,
                     const int* __restrict__ mel_len,
                     const unsigned char* __restrict__ src_mask,
                     const float* __restrict__ dtw_res,
                     float* __restrict__ out) {
    __shared__ float dl[B_];
    int t = threadIdx.x;
    if (t < B_) {
        float sum = 0.0f, cnt = 0.0f;
        for (int s = 0; s < S_; ++s) {
            bool m = src_mask[t * S_ + s] != 0;
            sum += m ? 0.0f : d[t * S_ + s];
            cnt += m ? 0.0f : 1.0f;
        }
        dl[t] = fabsf((float)mel_len[t] - sum) / cnt;
    }
    __syncthreads();
    if (t == 0) {
        float dls = 0.0f, m0 = 0.0f, m1 = 0.0f;
        for (int b = 0; b < B_; ++b) {
            dls += dl[b];
            m0  += dtw_res[b];
            m1  += dtw_res[B_ + b];
        }
        out[0] = 0.001f * (m0 / (float)B_);
        out[1] = 0.001f * (m1 / (float)B_);
        out[2] = dls / (float)B_;
    }
}

extern "C" void kernel_launch(void* const* d_in, const int* in_sizes, int n_in,
                              void* d_out, int out_size, void* d_ws, size_t ws_size,
                              hipStream_t stream) {
    const float*         d_arr       = (const float*)d_in[0];
    const int*           mel_len     = (const int*)d_in[1];
    const float*         mel         = (const float*)d_in[2];
    const float*         mel_postnet = (const float*)d_in[3];
    const float*         mel_target  = (const float*)d_in[4];
    const unsigned char* src_mask    = (const unsigned char*)d_in[5];
    const unsigned char* pred_mask   = (const unsigned char*)d_in[6];
    float* out = (float*)d_out;

    // workspace layout (16B-aligned sections):
    float* D       = (float*)d_ws;                               // 2*16*800*800 f32
    float* dtw_res = D + (size_t)2 * B_ * T_ * T_;               // 32 f32
    float* xn      = dtw_res + 32;                               // 2*16*800 f32
    float* yn      = xn + (size_t)2 * B_ * T_;                   // 16*800 f32
    _Float16* Xh   = (_Float16*)(yn + (size_t)B_ * T_);          // 2*16*800*96 f16
    _Float16* Yh   = Xh + (size_t)2 * B_ * T_ * KPAD;            // 16*800*96 f16

    // 3*16*800 rows, 4 rows (waves) per block
    prep_kernel<<<(3 * B_ * T_) / 4, 128, 0, stream>>>(
        mel, mel_postnet, mel_target, pred_mask, Xh, Yh, xn, yn);

    dim3 cgrid(T_ / TILE, T_ / TILE, 2 * B_);  // 25 x 25 x 32
    cost_kernel<<<cgrid, 128, 0, stream>>>(Xh, Yh, xn, yn, D);

    dtw_kernel<<<2 * B_, T_, 0, stream>>>(D, dtw_res);

    finalize_kernel<<<1, 32, 0, stream>>>(d_arr, mel_len, src_mask, dtw_res, out);
}
